// DepPairingLayer_50577534878508
// MI455X (gfx1250) — matrix operand order repaired and checked by reading, not compiled, not run on hardware
//
#include <hip/hip_runtime.h>

#define DEV __device__ __forceinline__

typedef __attribute__((ext_vector_type(16))) __bf16 v16bf;
typedef __attribute__((ext_vector_type(8)))  __bf16 v8bf;
typedef __attribute__((ext_vector_type(8)))  float  v8f;

// ---- problem sizes -------------------------------------------------------
constexpr int L_STEPS = 16;
constexpr int D_IN    = 832;
constexpr int H_DIM   = 384;
constexpr int DT_DIM  = 768;
constexpr int P_PAIRS = 8192;          // B*PB
constexpr int NG      = 4 * H_DIM;     // [iou | f] = 1536 gate columns
constexpr int DEC_IN  = 3 * H_DIM + 2 * DT_DIM;   // 2688
constexpr int DEC_H   = 512;
constexpr int DEC_OUT = 7;

constexpr int KT_W = D_IN / 32;        // 26 k-tiles for x@W
constexpr int KT_U = H_DIM / 32;       // 12 k-tiles for h@U
constexpr int NT_G = NG / 16;          // 96 n-tiles of gates
constexpr int KT1  = DEC_IN / 32;      // 84
constexpr int NT1  = DEC_H / 16;       // 32
constexpr int KT2  = DEC_H / 32;       // 16

DEV v8f wmma_bf16(v16bf a, v16bf b, v8f c) {
  return __builtin_amdgcn_wmma_f32_16x16x32_bf16(
      /*neg_a=*/false, a, /*neg_b=*/false, b,
      /*c_mod=*/(short)0, c, /*reuse_a=*/false, /*reuse_b=*/false);
}

// A operand (16x32 bf16, M x K) from row-major [m][k] memory with row stride
// ldk. Lane m=lane&15, half=lane>>4; elems 0-15 hold
// k = 16*half + 0..15 (two 16B chunks), matching packed B.
DEV v16bf load_A_tile(const __bf16* sm, int kt, int ldk, int lane) {
  const int m = lane & 15, half = lane >> 4;
  const __bf16* base = sm + m * ldk + kt * 32 + 16 * half;
  v8bf lo = *(const v8bf*)base;
  v8bf hi = *(const v8bf*)(base + 8);
  return __builtin_shufflevector(lo, hi, 0,1,2,3,4,5,6,7,8,9,10,11,12,13,14,15);
}

// B operand: weights pre-packed so each lane's 16 values are contiguous.
DEV v16bf load_B_blk(const __bf16* blk, int lane) {
  return *((const v16bf*)blk + lane);
}

DEV float sigmoidf_(float x) { return 1.f / (1.f + __expf(-x)); }

// ---- weight repack into WMMA-B block layout ------------------------------
// dst block (bk,bn): 512 bf16; lane l holds 16 contiguous elems where
// n = 16*bn + (l&15), k = 32*bk + 16*(l>>4) + e.  Columns [0,nA) from srcA,
// [nA, nA+nB) from srcB, rest zero.
__global__ void pack_b_kernel(const float* __restrict__ srcA, int nA,
                              const float* __restrict__ srcB, int nB,
                              int K, int Ntot, __bf16* __restrict__ dst) {
  const int KT = K / 32, NT = Ntot / 16;
  const int tid = blockIdx.x * blockDim.x + threadIdx.x;
  if (tid >= KT * NT * 32) return;
  const int l  = tid & 31;
  const int blk = tid >> 5;
  const int bk = blk % KT, bn = blk / KT;
  const int n  = 16 * bn + (l & 15);
  __bf16* out = dst + ((size_t)(bn * KT + bk)) * 512 + (size_t)l * 16;
#pragma unroll
  for (int e = 0; e < 16; ++e) {
    const int k = 32 * bk + 16 * (l >> 4) + e;
    float v = 0.f;
    if (n < nA)                      v = srcA[(size_t)k * nA + n];
    else if (srcB && (n - nA) < nB)  v = srcB[(size_t)k * nB + (n - nA)];
    out[e] = (__bf16)v;
  }
}

// ---- fused tree-LSTM direction (16 pairs per workgroup) ------------------
// dir==0: up scan, captures h at t==root_idx -> pair cols [0,384)
// dir==1: down scan over reversed sequence; h after first step (orig pos 15)
//         = end_h -> cols [768,1152); final h = start_h -> cols [384,768)
__global__ __launch_bounds__(256)
void lstm_fused_kernel(const float* __restrict__ x,      // (P, L, D)
                       const __bf16* __restrict__ Wblk,  // KT_W x NT_G blocks
                       const __bf16* __restrict__ Ublk,  // KT_U x NT_G blocks
                       const float* __restrict__ biou,   // 1152
                       const float* __restrict__ bfb,    // 384
                       const int* __restrict__ root_idx, // P
                       __bf16* __restrict__ pair,        // (P, DEC_IN)
                       int dir) {
  __shared__ __bf16 xA[16 * D_IN];    // 26.0 KB  x_t tile, row-major
  __shared__ __bf16 hA[16 * H_DIM];   // 12.0 KB  h_{t-1} tile, row-major
  __shared__ int    sroot[16];

  const int tid  = threadIdx.x;
  const int lane = tid & 31;
  const int wave = tid >> 5;
  const int nl   = lane & 15;
  const int mh   = lane >> 4;         // C-layout: element (m = r + 8*mh, n = nl)
  const int p0   = blockIdx.x * 16;

  for (int i = tid; i < 16 * H_DIM; i += 256) hA[i] = (__bf16)0.f;
  if (tid < 16) sroot[tid] = root_idx[p0 + tid];

  // wave w owns gate columns [48w, 48w+48): ntile g*24 + 3w + s  (g=i,o,u,f)
  int my_nt[12];
#pragma unroll
  for (int j = 0; j < 12; ++j) my_nt[j] = (j / 3) * 24 + wave * 3 + (j % 3);
  float bias_n[12];
#pragma unroll
  for (int j = 0; j < 12; ++j) {
    const int n = my_nt[j] * 16 + nl;
    bias_n[j] = (n < 3 * H_DIM) ? biou[n] : bfb[n - 3 * H_DIM];
  }

  v8f acc[12];
  v8f cc[3];
#pragma unroll
  for (int s = 0; s < 3; ++s)
#pragma unroll
    for (int r = 0; r < 8; ++r) cc[s][r] = 0.f;

  __syncthreads();

  for (int t = 0; t < L_STEPS; ++t) {
    const int tl = dir ? (L_STEPS - 1 - t) : t;
    // stage x_t tile (fp32 -> bf16)
    for (int i = tid; i < 16 * D_IN; i += 256) {
      const int m = i / D_IN, d = i % D_IN;
      xA[i] = (__bf16)x[((size_t)(p0 + m) * L_STEPS + tl) * D_IN + d];
    }
    __syncthreads();

#pragma unroll
    for (int j = 0; j < 12; ++j)
#pragma unroll
      for (int r = 0; r < 8; ++r) acc[j][r] = bias_n[j];

    // x_t @ [Wiou | Wf]
    for (int kt = 0; kt < KT_W; ++kt) {
      v16bf a = load_A_tile(xA, kt, D_IN, lane);
#pragma unroll
      for (int j = 0; j < 12; ++j) {
        v16bf b = load_B_blk(Wblk + ((size_t)(my_nt[j] * KT_W + kt)) * 512, lane);
        acc[j] = wmma_bf16(a, b, acc[j]);
      }
    }
    // h_{t-1} @ [Uiou | Uf]
    for (int kt = 0; kt < KT_U; ++kt) {
      v16bf a = load_A_tile(hA, kt, H_DIM, lane);
#pragma unroll
      for (int j = 0; j < 12; ++j) {
        v16bf b = load_B_blk(Ublk + ((size_t)(my_nt[j] * KT_U + kt)) * 512, lane);
        acc[j] = wmma_bf16(a, b, acc[j]);
      }
    }
    __syncthreads();   // everyone done reading hA before overwrite

    // gates + state update, all in this wave's own accumulators
#pragma unroll
    for (int s = 0; s < 3; ++s) {
      const int hcol = wave * 48 + s * 16 + nl;
#pragma unroll
      for (int r = 0; r < 8; ++r) {
        const float i_ = sigmoidf_(acc[0 + s][r]);
        const float o_ = sigmoidf_(acc[3 + s][r]);
        const float u_ = tanhf(acc[6 + s][r]);
        const float f_ = sigmoidf_(acc[9 + s][r]);
        const float c_ = i_ * u_ + f_ * cc[s][r];
        cc[s][r] = c_;
        const float h_ = o_ * tanhf(c_);
        const int m = r + 8 * mh;
        hA[m * H_DIM + hcol] = (__bf16)h_;
        if (dir == 0) {
          if (sroot[m] == tl)
            pair[(size_t)(p0 + m) * DEC_IN + hcol] = (__bf16)h_;           // root_h
        } else {
          if (t == 0)
            pair[(size_t)(p0 + m) * DEC_IN + 2 * H_DIM + hcol] = (__bf16)h_; // end_h
          if (t == L_STEPS - 1)
            pair[(size_t)(p0 + m) * DEC_IN + H_DIM + hcol] = (__bf16)h_;     // start_h
        }
      }
    }
    __syncthreads();   // hA fully written before next step consumes it
  }
}

// ---- span means (direct row sums; st+len+1 <= 503 < T) -------------------
__global__ __launch_bounds__(256)
void span_mean_kernel(const float* __restrict__ tok,  // (B, T, DT)
                      const int* __restrict__ st, const int* __restrict__ ln,
                      __bf16* __restrict__ pair, int colbase) {
  const int p = blockIdx.x;
  const int b = p >> 8;                       // PB = 256
  const int s = st[p], l = ln[p];
  const float* base = tok + ((size_t)b * 512 + s) * DT_DIM;
  const int c = threadIdx.x;
  float a0 = 0.f, a1 = 0.f, a2 = 0.f;
  for (int r = 0; r <= l; ++r) {
    const float* row = base + (size_t)r * DT_DIM;
    a0 += row[c]; a1 += row[c + 256]; a2 += row[c + 512];
  }
  const float inv = 1.f / (float)(l + 1);
  __bf16* out = pair + (size_t)p * DEC_IN + colbase;
  out[c]       = (__bf16)(a0 * inv);
  out[c + 256] = (__bf16)(a1 * inv);
  out[c + 512] = (__bf16)(a2 * inv);
}

// ---- decoder: tanh(pair@W1+b1)@W2+b2 ------------------------------------
__global__ __launch_bounds__(256)
void decoder_kernel(const __bf16* __restrict__ pair,
                    const __bf16* __restrict__ W1blk, const float* __restrict__ b1,
                    const __bf16* __restrict__ W2blk, const float* __restrict__ b2,
                    float* __restrict__ out) {
  __shared__ __bf16 sA[16 * DEC_IN];   // 84.0 KB
  __shared__ __bf16 sH[16 * DEC_H];    // 16.0 KB
  const int tid = threadIdx.x, lane = tid & 31, wave = tid >> 5;
  const int nl = lane & 15, mh = lane >> 4;
  const int p0 = blockIdx.x * 16;

  // stage 16 full pair rows (contiguous) into LDS
  const uint4* src = (const uint4*)(pair + (size_t)p0 * DEC_IN);
  uint4* dst = (uint4*)sA;
  for (int i = tid; i < 16 * DEC_IN * 2 / 16; i += 256) dst[i] = src[i];
  __syncthreads();

  v8f acc[4];
#pragma unroll
  for (int j = 0; j < 4; ++j) {
    const float bv = b1[(wave * 4 + j) * 16 + nl];
#pragma unroll
    for (int r = 0; r < 8; ++r) acc[j][r] = bv;
  }
  for (int kt = 0; kt < KT1; ++kt) {
    if (kt + 1 < KT1)
      __builtin_prefetch(W1blk + ((size_t)(wave * 4) * KT1 + kt + 1) * 512, 0, 0);
    v16bf a = load_A_tile(sA, kt, DEC_IN, lane);
#pragma unroll
    for (int j = 0; j < 4; ++j) {
      v16bf b = load_B_blk(W1blk + ((size_t)((wave * 4 + j) * KT1 + kt)) * 512, lane);
      acc[j] = wmma_bf16(a, b, acc[j]);
    }
  }
  // tanh -> LDS (row-major 16 x 512 bf16)
#pragma unroll
  for (int j = 0; j < 4; ++j) {
    const int n = (wave * 4 + j) * 16 + nl;
#pragma unroll
    for (int r = 0; r < 8; ++r)
      sH[(r + 8 * mh) * DEC_H + n] = (__bf16)tanhf(acc[j][r]);
  }
  __syncthreads();

  if (wave == 0) {                      // wave-uniform: EXEC stays all-ones
    v8f a2;
    const float bv = (nl < DEC_OUT) ? b2[nl] : 0.f;
#pragma unroll
    for (int r = 0; r < 8; ++r) a2[r] = bv;
    for (int kt = 0; kt < KT2; ++kt) {
      v16bf a = load_A_tile(sH, kt, DEC_H, lane);
      v16bf b = load_B_blk(W2blk + (size_t)kt * 512, lane);
      a2 = wmma_bf16(a, b, a2);
    }
    if (nl < DEC_OUT) {
#pragma unroll
      for (int r = 0; r < 8; ++r)
        out[(size_t)(p0 + r + 8 * mh) * DEC_OUT + nl] = a2[r];
    }
  }
}

// ---- host side -----------------------------------------------------------
extern "C" void kernel_launch(void* const* d_in, const int* in_sizes, int n_in,
                              void* d_out, int out_size, void* d_ws, size_t ws_size,
                              hipStream_t stream) {
  (void)in_sizes; (void)n_in; (void)out_size; (void)ws_size;
  const float* node_embs  = (const float*)d_in[0];
  const float* token_embs = (const float*)d_in[1];
  const int*   root_idx   = (const int*)d_in[2];
  const int*   p1_st      = (const int*)d_in[3];
  const int*   p1_len     = (const int*)d_in[4];
  const int*   p2_st      = (const int*)d_in[5];
  const int*   p2_len     = (const int*)d_in[6];
  const float* Wiou_u = (const float*)d_in[7];
  const float* Uiou_u = (const float*)d_in[8];
  const float* biou_u = (const float*)d_in[9];
  const float* Wf_u   = (const float*)d_in[10];
  const float* Uf_u   = (const float*)d_in[11];
  const float* bf_u   = (const float*)d_in[12];
  const float* Wiou_d = (const float*)d_in[13];
  const float* Uiou_d = (const float*)d_in[14];
  const float* biou_d = (const float*)d_in[15];
  const float* Wf_d   = (const float*)d_in[16];
  const float* Uf_d   = (const float*)d_in[17];
  const float* bf_d   = (const float*)d_in[18];
  const float* W1     = (const float*)d_in[19];
  const float* b1     = (const float*)d_in[20];
  const float* W2     = (const float*)d_in[21];
  const float* b2     = (const float*)d_in[22];

  // workspace carve (all 256B-aligned by construction)
  char* w = (char*)d_ws;
  __bf16* WcatU = (__bf16*)w; w += (size_t)KT_W * NT_G * 512 * 2;
  __bf16* UcatU = (__bf16*)w; w += (size_t)KT_U * NT_G * 512 * 2;
  __bf16* WcatD = (__bf16*)w; w += (size_t)KT_W * NT_G * 512 * 2;
  __bf16* UcatD = (__bf16*)w; w += (size_t)KT_U * NT_G * 512 * 2;
  __bf16* W1blk = (__bf16*)w; w += (size_t)KT1 * NT1 * 512 * 2;
  __bf16* W2blk = (__bf16*)w; w += (size_t)KT2 * 1 * 512 * 2;
  __bf16* pair  = (__bf16*)w; w += (size_t)P_PAIRS * DEC_IN * 2;

  auto blocks = [](int total) { return (total + 255) / 256; };

  // 1) pack weights to bf16 WMMA-B block layout
  pack_b_kernel<<<blocks(KT_W * NT_G * 32), 256, 0, stream>>>(
      Wiou_u, 3 * H_DIM, Wf_u, H_DIM, D_IN, NG, WcatU);
  pack_b_kernel<<<blocks(KT_U * NT_G * 32), 256, 0, stream>>>(
      Uiou_u, 3 * H_DIM, Uf_u, H_DIM, H_DIM, NG, UcatU);
  pack_b_kernel<<<blocks(KT_W * NT_G * 32), 256, 0, stream>>>(
      Wiou_d, 3 * H_DIM, Wf_d, H_DIM, D_IN, NG, WcatD);
  pack_b_kernel<<<blocks(KT_U * NT_G * 32), 256, 0, stream>>>(
      Uiou_d, 3 * H_DIM, Uf_d, H_DIM, H_DIM, NG, UcatD);
  pack_b_kernel<<<blocks(KT1 * NT1 * 32), 256, 0, stream>>>(
      W1, DEC_H, nullptr, 0, DEC_IN, DEC_H, W1blk);
  pack_b_kernel<<<blocks(KT2 * 1 * 32), 256, 0, stream>>>(
      W2, DEC_OUT, nullptr, 0, DEC_H, 16, W2blk);

  // 2) fused bidirectional LSTM (root_h / start_h / end_h -> pair staging)
  lstm_fused_kernel<<<P_PAIRS / 16, 256, 0, stream>>>(
      node_embs, WcatU, UcatU, biou_u, bf_u, root_idx, pair, 0);
  lstm_fused_kernel<<<P_PAIRS / 16, 256, 0, stream>>>(
      node_embs, WcatD, UcatD, biou_d, bf_d, root_idx, pair, 1);

  // 3) span means -> pair staging
  span_mean_kernel<<<P_PAIRS, 256, 0, stream>>>(
      token_embs, p1_st, p1_len, pair, 3 * H_DIM);
  span_mean_kernel<<<P_PAIRS, 256, 0, stream>>>(
      token_embs, p2_st, p2_len, pair, 3 * H_DIM + DT_DIM);

  // 4) fused decoder -> logits
  decoder_kernel<<<P_PAIRS / 16, 256, 0, stream>>>(
      pair, W1blk, b1, W2blk, b2, (float*)d_out);
}
